// Attention_66142496358763
// MI455X (gfx1250) — matrix operand order repaired and checked by reading, not compiled, not run on hardware
//
#include <hip/hip_runtime.h>

typedef __attribute__((ext_vector_type(16))) _Float16 v16h;
typedef __attribute__((ext_vector_type(4)))  _Float16 v4h;
typedef __attribute__((ext_vector_type(8)))  float    v8f;

#define S_LEN 2048
#define DIM   4096
#define NH    32
#define NKV   8
#define HD    128

// ---------------------------------------------------------------------------
// Generic GEMM: C[M,N](f32) = A[M,K](f32) @ B[K,N](f32), f16 WMMA compute.
// N,K are template params so all strides fold into immediate offsets.
// 256 threads = 8 waves (4x2 wave grid, each wave owns a 32x64 output tile).
// B tile staged TRANSPOSED in LDS so B-operand fills are 2x ds_load_b128.
// ---------------------------------------------------------------------------
#define BM 128
#define BN 128
#define BK 32
#define LDB (BK + 8)   // 40 halfs = 80 B stride: 16-B aligned, bank-spread

template <int N, int K>
__global__ __launch_bounds__(256) void gemm_f16w(const float* __restrict__ A,
                                                 const float* __restrict__ B,
                                                 float* __restrict__ C) {
  __shared__ _Float16 As[BM * BK];        // 8 KB, row-major [m][k]
  __shared__ _Float16 BsT[BN * LDB];      // 10 KB, transposed [n][k]
  const int tid  = threadIdx.x;
  const int lane = tid & 31;
  const int l16  = lane & 15;
  const int lhi  = lane >> 4;
  const int wave = tid >> 5;
  const int wm   = (wave >> 1) * 32;  // 0,32,64,96
  const int wn   = (wave & 1) * 64;   // 0,64
  const long bm0 = (long)blockIdx.y * BM;
  const long bn0 = (long)blockIdx.x * BN;

  const int trow = tid >> 3;          // A staging: row 0..31 (+32*p)
  const int tc4  = (tid & 7) << 2;    // A staging: k offset
  const int brow = tid >> 5;          // B staging: k row 0..7 (+8*p)
  const int bc4  = lane << 2;         // B staging: n offset

  v8f acc[2][4];
#pragma unroll
  for (int i = 0; i < 2; ++i)
#pragma unroll
    for (int j = 0; j < 4; ++j)
#pragma unroll
      for (int r = 0; r < 8; ++r) acc[i][j][r] = 0.0f;

  for (int kb = 0; kb < K; kb += BK) {
    // Stage A tile (128x32) fp32 -> f16 LDS, packed b64 stores
#pragma unroll
    for (int p = 0; p < 4; ++p) {
      const int r = p * 32 + trow;
      const float4 v = *(const float4*)(A + (bm0 + r) * K + kb + tc4);
      v4h pk = {(_Float16)v.x, (_Float16)v.y, (_Float16)v.z, (_Float16)v.w};
      *(v4h*)(As + r * BK + tc4) = pk;
    }
    // Stage B tile (32x128) transposed into BsT[n][k]
#pragma unroll
    for (int p = 0; p < 4; ++p) {
      const int r = p * 8 + brow;  // k index
      const float4 v = *(const float4*)(B + (long)(kb + r) * N + bn0 + bc4);
      BsT[(bc4 + 0) * LDB + r] = (_Float16)v.x;
      BsT[(bc4 + 1) * LDB + r] = (_Float16)v.y;
      BsT[(bc4 + 2) * LDB + r] = (_Float16)v.z;
      BsT[(bc4 + 3) * LDB + r] = (_Float16)v.w;
    }
    // Prefetch next tiles into cache (global_prefetch_b8)
    if (kb + BK < K) {
      __builtin_prefetch(A + (bm0 + trow) * K + kb + BK + tc4, 0, 0);
      __builtin_prefetch(B + (long)(kb + BK + brow) * N + bn0 + bc4, 0, 0);
    }
    __syncthreads();

    // A fill (16x32 f16), contiguous k order used here, same as B: k = h + 16*(lane>=16), m = lane&15
    v16h a[2];
#pragma unroll
    for (int ci = 0; ci < 2; ++ci) {
      const int mrow = wm + ci * 16 + l16;
#pragma unroll
      for (int h = 0; h < 16; ++h) {
        const int k = h + (lhi << 4);
        a[ci][h] = As[mrow * BK + k];
      }
    }
    // B fill (32x16 f16), same contiguous k order as A: k = h + 16*(lane>=16), n = lane&15 -> contiguous
#pragma unroll
    for (int cj = 0; cj < 4; ++cj) {
      v16h b;
      const int ncol = wn + cj * 16 + l16;
#pragma unroll
      for (int h = 0; h < 16; ++h) {
        b[h] = BsT[ncol * LDB + h + (lhi << 4)];
      }
#pragma unroll
      for (int ci = 0; ci < 2; ++ci)
        acc[ci][cj] = __builtin_amdgcn_wmma_f32_16x16x32_f16(
            false, a[ci], false, b, (short)0, acc[ci][cj], false, false);
    }
    __syncthreads();
  }

  // C layout: row = r + 8*(lane>=16), col = lane&15
#pragma unroll
  for (int ci = 0; ci < 2; ++ci)
#pragma unroll
    for (int cj = 0; cj < 4; ++cj)
#pragma unroll
      for (int r = 0; r < 8; ++r) {
        const long row = bm0 + wm + ci * 16 + r + (lhi << 3);
        const long col = bn0 + wn + cj * 16 + l16;
        C[row * N + col] = acc[ci][cj][r];
      }
}

// ---------------------------------------------------------------------------
// RoPE (in-place, fp32). Layout t: [S, NHEADS, 128]; cos/sin: [S, 64].
// ---------------------------------------------------------------------------
template <int NHEADS>
__global__ void rope_kernel(float* __restrict__ t, const float* __restrict__ fc,
                            const float* __restrict__ fs) {
  const int idx = blockIdx.x * blockDim.x + threadIdx.x;
  if (idx >= S_LEN * NHEADS * 64) return;
  const int d2 = idx & 63;
  const int h  = (idx >> 6) % NHEADS;      // NHEADS is a power of two
  const int s  = idx / (NHEADS << 6);
  const float c  = fc[s * 64 + d2];
  const float sn = fs[s * 64 + d2];
  const long base = ((long)s * NHEADS + h) * HD + (d2 << 1);
  const float t0 = t[base];
  const float t1 = t[base + 1];
  t[base]     = t0 * c - t1 * sn;
  t[base + 1] = t0 * sn + t1 * c;
}

// ---------------------------------------------------------------------------
// Flash attention with GQA. Block = 128 threads = 4 waves; each wave owns 16
// query rows. Grid = (S/64, NH). Streams 64-row K/V tiles through LDS.
// K staged row-major (QK^T B-fill walks head_dim: contiguous);
// V staged transposed (PV B-fill walks kv rows: contiguous).
// ---------------------------------------------------------------------------
#define ABM 64          // query rows per block
#define ABN 64          // kv rows per tile
#define LDV (ABN + 8)   // 72 halfs = 144 B stride: 16-B aligned, bank-spread

__global__ __launch_bounds__(128) void attn_kernel(const float* __restrict__ Q,
                                                   const float* __restrict__ Km,
                                                   const float* __restrict__ Vm,
                                                   float* __restrict__ O) {
  __shared__ _Float16 Ks[ABN * HD];       // 16 KB, [kv][hd]
  __shared__ _Float16 VsT[HD * LDV];      // 18 KB, [hd][kv]
  __shared__ _Float16 Ps[4][16 * ABN];    // 8 KB (per-wave P staging)

  const int tid  = threadIdx.x;
  const int lane = tid & 31;
  const int l16  = lane & 15;
  const int lhi  = lane >> 4;
  const int wave = tid >> 5;
  const int head = blockIdx.y;
  const int kvh  = head >> 2;             // GQA: 32 heads / 8 kv heads
  const int q0   = blockIdx.x * ABM + wave * 16;
  const float scale = 0.08838834764831845f;  // 1/sqrt(128)

  // Q in WMMA A-layout (4 chunks of K=32 spanning head_dim)
  v16h qa[4];
  {
    const float* qrow = Q + (long)(q0 + l16) * DIM + (long)head * HD;
#pragma unroll
    for (int c = 0; c < 4; ++c)
#pragma unroll
      for (int h = 0; h < 16; ++h) {
        const int k = c * 32 + h + (lhi << 4);
        qa[c][h] = (_Float16)qrow[k];
      }
  }

  float m_i[8], l_i[8];
  v8f acc[8];
#pragma unroll
  for (int r = 0; r < 8; ++r) { m_i[r] = -1.0e30f; l_i[r] = 0.0f; }
#pragma unroll
  for (int ob = 0; ob < 8; ++ob)
#pragma unroll
    for (int r = 0; r < 8; ++r) acc[ob][r] = 0.0f;

  const int r0 = tid >> 5;
  const int c4 = lane << 2;

  for (int kb = 0; kb < S_LEN; kb += ABN) {
    // Cooperative stage: K row-major (packed b64), V transposed (b16 scatter)
#pragma unroll
    for (int p = 0; p < 16; ++p) {
      const int r = p * 4 + r0;
      const float4 kv = *(const float4*)(Km + (long)(kb + r) * (NKV * HD) + (long)kvh * HD + c4);
      v4h pk = {(_Float16)kv.x, (_Float16)kv.y, (_Float16)kv.z, (_Float16)kv.w};
      *(v4h*)(Ks + r * HD + c4) = pk;
      const float4 vv = *(const float4*)(Vm + (long)(kb + r) * (NKV * HD) + (long)kvh * HD + c4);
      VsT[(c4 + 0) * LDV + r] = (_Float16)vv.x;
      VsT[(c4 + 1) * LDV + r] = (_Float16)vv.y;
      VsT[(c4 + 2) * LDV + r] = (_Float16)vv.z;
      VsT[(c4 + 3) * LDV + r] = (_Float16)vv.w;
    }
    // Prefetch next K/V tile rows
    if (kb + ABN < S_LEN) {
      __builtin_prefetch(Km + (long)(kb + ABN + r0) * (NKV * HD) + (long)kvh * HD + c4, 0, 0);
      __builtin_prefetch(Vm + (long)(kb + ABN + r0) * (NKV * HD) + (long)kvh * HD + c4, 0, 0);
    }
    __syncthreads();

    // S = scale * Q @ K^T   (16 x 64 per wave, 4 n-blocks x 4 k-chunks)
    v8f s[4];
#pragma unroll
    for (int nb = 0; nb < 4; ++nb) {
      v8f sc = {};
#pragma unroll
      for (int c = 0; c < 4; ++c) {
        v16h b;  // B[k][n] = K[kv_row = nb*16+n][hd = c*32+k] -> contiguous in h
#pragma unroll
        for (int h = 0; h < 16; ++h) {
          b[h] = Ks[(nb * 16 + l16) * HD + c * 32 + h + (lhi << 4)];
        }
        sc = __builtin_amdgcn_wmma_f32_16x16x32_f16(false, qa[c], false, b,
                                                    (short)0, sc, false, false);
      }
#pragma unroll
      for (int r = 0; r < 8; ++r) sc[r] *= scale;
      s[nb] = sc;
    }

    // Online softmax: rows live in 16-lane halves -> xor masks 1,2,4,8
    float alpha[8];
#pragma unroll
    for (int r = 0; r < 8; ++r) {
      float t = fmaxf(fmaxf(s[0][r], s[1][r]), fmaxf(s[2][r], s[3][r]));
      t = fmaxf(t, __shfl_xor(t, 1, 32));
      t = fmaxf(t, __shfl_xor(t, 2, 32));
      t = fmaxf(t, __shfl_xor(t, 4, 32));
      t = fmaxf(t, __shfl_xor(t, 8, 32));
      const float mnew = fmaxf(m_i[r], t);
      alpha[r] = __expf(m_i[r] - mnew);
      float su = 0.0f;
#pragma unroll
      for (int nb = 0; nb < 4; ++nb) {
        const float p = __expf(s[nb][r] - mnew);
        s[nb][r] = p;
        su += p;
      }
      su += __shfl_xor(su, 1, 32);
      su += __shfl_xor(su, 2, 32);
      su += __shfl_xor(su, 4, 32);
      su += __shfl_xor(su, 8, 32);
      l_i[r] = l_i[r] * alpha[r] + su;
      m_i[r] = mnew;
    }
#pragma unroll
    for (int ob = 0; ob < 8; ++ob)
#pragma unroll
      for (int r = 0; r < 8; ++r) acc[ob][r] *= alpha[r];

    // C-layout P -> LDS -> A-layout (per-wave region; same-wave DS is in-order)
#pragma unroll
    for (int nb = 0; nb < 4; ++nb)
#pragma unroll
      for (int r = 0; r < 8; ++r)
        Ps[wave][(r + (lhi << 3)) * ABN + nb * 16 + l16] = (_Float16)s[nb][r];

    v16h pa[2];
#pragma unroll
    for (int c = 0; c < 2; ++c)
#pragma unroll
      for (int h = 0; h < 16; ++h) {
        const int k = c * 32 + h + (lhi << 4);
        pa[c][h] = Ps[wave][l16 * ABN + k];
      }

    // O += P @ V   (8 hd-blocks x 2 k-chunks); VsT fill contiguous in h
#pragma unroll
    for (int ob = 0; ob < 8; ++ob)
#pragma unroll
      for (int c = 0; c < 2; ++c) {
        v16h bv;  // B[k][n] = V[kv_row = c*32+k][hd = ob*16+n]
#pragma unroll
        for (int h = 0; h < 16; ++h) {
          bv[h] = VsT[(ob * 16 + l16) * LDV + c * 32 + h + (lhi << 4)];
        }
        acc[ob] = __builtin_amdgcn_wmma_f32_16x16x32_f16(false, pa[c], false, bv,
                                                         (short)0, acc[ob], false, false);
      }
    __syncthreads();
  }

  // Normalize and write attn_out[row][head*128 + col] (fp32)
#pragma unroll
  for (int ob = 0; ob < 8; ++ob)
#pragma unroll
    for (int r = 0; r < 8; ++r) {
      const long row = q0 + r + (lhi << 3);
      const long col = (long)head * HD + ob * 16 + l16;
      O[row * DIM + col] = acc[ob][r] / l_i[r];
    }
}

// ---------------------------------------------------------------------------
extern "C" void kernel_launch(void* const* d_in, const int* in_sizes, int n_in,
                              void* d_out, int out_size, void* d_ws, size_t ws_size,
                              hipStream_t stream) {
  const float* x  = (const float*)d_in[0];
  const float* fc = (const float*)d_in[1];
  const float* fs = (const float*)d_in[2];
  const float* wq = (const float*)d_in[3];
  const float* wk = (const float*)d_in[4];
  const float* wv = (const float*)d_in[5];
  const float* wo = (const float*)d_in[6];
  float* out = (float*)d_out;

  float* Qf = (float*)d_ws;                       // S * DIM
  float* Kf = Qf + (size_t)S_LEN * DIM;           // S * NKV*HD
  float* Vf = Kf + (size_t)S_LEN * NKV * HD;      // S * NKV*HD
  float* AO = Vf + (size_t)S_LEN * NKV * HD;      // S * DIM

  const dim3 blk(256);
  gemm_f16w<DIM, DIM><<<dim3(DIM / BN, S_LEN / BM), blk, 0, stream>>>(x, wq, Qf);
  gemm_f16w<NKV * HD, DIM><<<dim3((NKV * HD) / BN, S_LEN / BM), blk, 0, stream>>>(x, wk, Kf);
  gemm_f16w<NKV * HD, DIM><<<dim3((NKV * HD) / BN, S_LEN / BM), blk, 0, stream>>>(x, wv, Vf);

  rope_kernel<NH><<<(S_LEN * NH * 64 + 255) / 256, 256, 0, stream>>>(Qf, fc, fs);
  rope_kernel<NKV><<<(S_LEN * NKV * 64 + 255) / 256, 256, 0, stream>>>(Kf, fc, fs);

  attn_kernel<<<dim3(S_LEN / ABM, NH), dim3(128), 0, stream>>>(Qf, Kf, Vf, AO);

  gemm_f16w<DIM, DIM><<<dim3(DIM / BN, S_LEN / BM), blk, 0, stream>>>(AO, wo, out);
}
